// RoPEAttentionFused_42949672960536
// MI455X (gfx1250) — matrix operand order repaired and checked by reading, not compiled, not run on hardware
//
#include <hip/hip_runtime.h>
#include <hip/hip_bf16.h>

// Problem constants (match reference)
#define BQ 4
#define SQ 2048
#define DQ 2048
#define HQ 16
#define HDQ 128

typedef __attribute__((ext_vector_type(16))) _Float16 v16h;
typedef __attribute__((ext_vector_type(8)))  _Float16 v8h;
typedef __attribute__((ext_vector_type(8)))  float    v8f;

static __device__ __forceinline__ v16h cat16(v8h lo, v8h hi) {
  return __builtin_shufflevector(lo, hi, 0,1,2,3,4,5,6,7,8,9,10,11,12,13,14,15);
}
static __device__ __forceinline__ v8f wmma_f16(v16h a, v16h b, v8f c) {
  // D = A*B + C, f32 accumulate, 16x16x32
  return __builtin_amdgcn_wmma_f32_16x16x32_f16(false, a, false, b, (short)0, c, false, false);
}

// gfx1250 async Global->LDS copy (16B per lane), tracked by ASYNCcnt.
// VDST operand = LDS byte offset (low 32 bits of the generic shared pointer,
// per the LDS aperture mapping), VADDR = 64-bit global address.
static __device__ __forceinline__ void async_ld16(void* lds, const void* g) {
  asm volatile("global_load_async_to_lds_b128 %0, %1, off"
               :
               : "v"((unsigned)(unsigned long long)lds), "v"(g)
               : "memory");
}
static __device__ __forceinline__ void wait_async0() {
  asm volatile("s_wait_asynccnt 0x0" ::: "memory");
}

// ---------------------------------------------------------------------------
// Kernel 0: fp32 -> f16 cast (grid-strided, 8 elements / thread / step)
// ---------------------------------------------------------------------------
__global__ void cast_f32_to_f16(const float* __restrict__ src,
                                _Float16* __restrict__ dst, int n8) {
  int i = blockIdx.x * blockDim.x + threadIdx.x;
  int stride = gridDim.x * blockDim.x;
  for (; i < n8; i += stride) {
    const float4* s = (const float4*)src + (size_t)i * 2;
    float4 f0 = s[0], f1 = s[1];
    v8h h;
    h[0] = (_Float16)f0.x; h[1] = (_Float16)f0.y;
    h[2] = (_Float16)f0.z; h[3] = (_Float16)f0.w;
    h[4] = (_Float16)f1.x; h[5] = (_Float16)f1.y;
    h[6] = (_Float16)f1.z; h[7] = (_Float16)f1.w;
    *((v8h*)dst + i) = h;
  }
}

// ---------------------------------------------------------------------------
// Kernel 1: QKV GEMM (x[8192,2048] @ wqkv^T[2048,6144]) with fused RoPE.
// Block tile 128(M) x 128(N); 8 waves in 2x4 grid; each wave 64x32 (8 WMMA
// accumulator tiles). N-tile of 128 == exactly one head of one of q/k/v, so
// RoPE pairing (hd, hd+64) stays inside the block: stage fp32 C in LDS,
// rotate, emit f16. q,k -> [B,H,S,HD]; v -> transposed [B,H,HD,S].
// ---------------------------------------------------------------------------
__global__ __launch_bounds__(256) void qkv_rope_kernel(
    const _Float16* __restrict__ xh,    // [B*S, D]
    const _Float16* __restrict__ wh,    // [3D, D]  (N x K row-major)
    _Float16* __restrict__ qh,          // [B,H,S,HD]
    _Float16* __restrict__ kh,          // [B,H,S,HD]
    _Float16* __restrict__ vth)         // [B,H,HD,S]
{
  __shared__ __align__(16) float smemf[128 * 128];   // 64 KB, reused
  _Float16* As = (_Float16*)smemf;                   // [128][32] f16
  _Float16* Bs = As + 128 * 32;                      // [128][32] f16

  const int t = threadIdx.x;
  const int lane = t & 31, w = t >> 5;
  const int wm = w >> 2, wn = w & 3;
  const int m0 = blockIdx.y * 128;
  const int n0 = blockIdx.x * 128;
  const int mla = lane & 15, half = lane >> 4;
  const int kb8 = half * 8, koff16 = half * 16;
  const int arow = t >> 1, acol = (t & 1) * 16;

  v8f acc[4][2] = {};

  for (int kt = 0; kt < DQ; kt += 32) {
    // async staging: 16B/lane direct Global->LDS, no VGPR round-trip
    const _Float16* gp = xh + (size_t)(m0 + arow) * DQ + kt + acol;
    async_ld16(As + arow * 32 + acol,     gp);
    async_ld16(As + arow * 32 + acol + 8, gp + 8);
    const _Float16* gq = wh + (size_t)(n0 + arow) * DQ + kt + acol;
    async_ld16(Bs + arow * 32 + acol,     gq);
    async_ld16(Bs + arow * 32 + acol + 8, gq + 8);
    if (kt + 32 < DQ) {                 // gfx1250 global_prefetch of next tile
      __builtin_prefetch(gp + 32, 0, 1);
      __builtin_prefetch(gq + 32, 0, 1);
    }
    wait_async0();
    __syncthreads();

    v16h af[4], bf[2];
#pragma unroll
    for (int i = 0; i < 4; ++i) {
      const _Float16* p = As + (wm * 64 + i * 16 + mla) * 32;
      af[i] = cat16(*(const v8h*)(p + kb8), *(const v8h*)(p + kb8 + 16));
    }
#pragma unroll
    for (int j = 0; j < 2; ++j) {
      const _Float16* p = Bs + (wn * 32 + j * 16 + mla) * 32 + kb8;
      bf[j] = cat16(*(const v8h*)p, *(const v8h*)(p + 16));
    }
#pragma unroll
    for (int i = 0; i < 4; ++i)
#pragma unroll
      for (int j = 0; j < 2; ++j)
        acc[i][j] = wmma_f16(af[i], bf[j], acc[i][j]);
    __syncthreads();
  }

  // dump fp32 C tile to LDS (documented C layout: VGPR r -> row r (+8 hi half))
#pragma unroll
  for (int i = 0; i < 4; ++i)
#pragma unroll
    for (int j = 0; j < 2; ++j)
#pragma unroll
      for (int r = 0; r < 8; ++r) {
        int ml = wm * 64 + i * 16 + r + half * 8;
        int nl = wn * 32 + j * 16 + mla;
        smemf[ml * 128 + nl] = acc[i][j][r];
      }
  __syncthreads();

  const int sel = n0 >> 11;             // 0=q 1=k 2=v
  const int h   = (n0 & 2047) >> 7;     // head index
  if (sel < 2) {
    _Float16* dst = (sel == 0) ? qh : kh;
    for (int idx = t; idx < 128 * 64; idx += 256) {
      int r = idx >> 6, i = idx & 63;
      int mg = m0 + r;
      int b = mg >> 11, s = mg & 2047;
      float c1 = smemf[r * 128 + i];
      float c2 = smemf[r * 128 + i + 64];
      // inv_freq[i] = 10000^(-i/64) = exp2(-i * log2(10000)/64)
      float invf = exp2f(-(float)i * (13.28771237954945f / 64.0f));
      float fr = (float)s * invf;
      float sn, cs;
      __sincosf(fr, &sn, &cs);
      size_t base = (((size_t)b * HQ + h) * SQ + s) * HDQ;
      dst[base + i]      = (_Float16)(c1 * cs - c2 * sn);
      dst[base + i + 64] = (_Float16)(c2 * cs + c1 * sn);
    }
  } else {
    for (int idx = t; idx < 128 * 128; idx += 256) {
      int r = idx >> 7, c = idx & 127;
      int mg = m0 + r;
      int b = mg >> 11, s = mg & 2047;
      vth[(((size_t)b * HQ + h) * HDQ + c) * SQ + s] = (_Float16)smemf[r * 128 + c];
    }
  }
}

// ---------------------------------------------------------------------------
// Kernel 2: causal flash attention. One block = (b,h, 128 query rows);
// 8 waves, 16 query rows each, streaming 32-key tiles through LDS.
// ---------------------------------------------------------------------------
__global__ __launch_bounds__(256) void attn_kernel(
    const _Float16* __restrict__ qh,    // [B,H,S,HD]
    const _Float16* __restrict__ kh,    // [B,H,S,HD]
    const _Float16* __restrict__ vth,   // [B,H,HD,S]
    _Float16* __restrict__ oh)          // [B,S,H*HD]
{
  __shared__ __align__(16) _Float16 klds[32 * 128];   // K tile [key][hd]
  __shared__ __align__(16) _Float16 vlds[128 * 32];   // V^T tile [hd][key]
  __shared__ __align__(16) _Float16 plds[8 * 16 * 32];// per-wave P transpose

  const int t = threadIdx.x, lane = t & 31, w = t >> 5;
  const int mla = lane & 15, half = lane >> 4;
  const int kb8 = half * 8, koff16 = half * 16;
  const int bh = blockIdx.y;            // b*H + h
  const int b = bh >> 4, h = bh & 15;
  const int q0 = blockIdx.x * 128;
  const int qw = q0 + w * 16;           // this wave's first query row

  const _Float16* qp = qh  + ((size_t)bh * SQ) * HDQ;
  const _Float16* kp = kh  + ((size_t)bh * SQ) * HDQ;
  const _Float16* vp = vth + ((size_t)bh * HDQ) * SQ;

  // Q fragments for this wave's 16 rows, all 4 K-chunks, kept in registers.
  v16h qf[4];
#pragma unroll
  for (int c = 0; c < 4; ++c) {
    const _Float16* p = qp + (size_t)(qw + mla) * HDQ + c * 32 + kb8;
    qf[c] = cat16(*(const v8h*)p, *(const v8h*)(p + 16));
  }

  v8f o[8] = {};
  float mI[8], lI[8];
#pragma unroll
  for (int r = 0; r < 8; ++r) { mI[r] = -1e30f; lI[r] = 0.f; }

  const float scale = 0.08838834764831845f;   // 1/sqrt(128)
  const int numkt = (q0 >> 5) + 4;            // keys up to q0+127

  for (int kt = 0; kt < numkt; ++kt) {
    const int kbase = kt * 32;
    {
      int row = t >> 3, col = (t & 7) * 16;   // K tile 32x128
      const _Float16* g = kp + (size_t)(kbase + row) * HDQ + col;
      async_ld16(klds + row * 128 + col,     g);
      async_ld16(klds + row * 128 + col + 8, g + 8);
      int vrow = t >> 1, vcol = (t & 1) * 16; // V^T tile 128x32
      const _Float16* gv = vp + (size_t)vrow * SQ + kbase + vcol;
      async_ld16(vlds + vrow * 32 + vcol,     gv);
      async_ld16(vlds + vrow * 32 + vcol + 8, gv + 8);
    }
    wait_async0();
    __syncthreads();

    if (kbase <= qw + 15) {             // wave-uniform: EXEC stays all-ones
      // scores = q @ k^T for 32 keys (2 N-tiles x 4 K-chunks)
      v8f sc[2] = {};
#pragma unroll
      for (int c = 0; c < 4; ++c)
#pragma unroll
        for (int j = 0; j < 2; ++j) {
          const _Float16* p = klds + (j * 16 + mla) * 128 + c * 32 + kb8;
          v16h bf = cat16(*(const v8h*)p, *(const v8h*)(p + 16));
          sc[j] = wmma_f16(qf[c], bf, sc[j]);
        }

      const bool needMask = (kbase + 31) > qw;
      float pr[2][8];
#pragma unroll
      for (int j = 0; j < 2; ++j)
#pragma unroll
        for (int r = 0; r < 8; ++r) {
          float v = sc[j][r] * scale;
          if (needMask) {
            int key   = kbase + j * 16 + mla;
            int query = qw + r + half * 8;
            if (key > query) v = -1e30f;
          }
          pr[j][r] = v;
        }

      // online softmax; C rows live in 16-lane halves -> xor masks 8..1
      _Float16* pw = plds + w * (16 * 32);
#pragma unroll
      for (int r = 0; r < 8; ++r) {
        float v = fmaxf(pr[0][r], pr[1][r]);
#pragma unroll
        for (int off = 8; off >= 1; off >>= 1)
          v = fmaxf(v, __shfl_xor(v, off, 32));
        float mnew = fmaxf(mI[r], v);
        float corr = __expf(mI[r] - mnew);
        mI[r] = mnew;
        float p0 = __expf(pr[0][r] - mnew);
        float p1 = __expf(pr[1][r] - mnew);
        float s = p0 + p1;
#pragma unroll
        for (int off = 8; off >= 1; off >>= 1)
          s += __shfl_xor(s, off, 32);
        lI[r] = lI[r] * corr + s;
#pragma unroll
        for (int nt = 0; nt < 8; ++nt)
          o[nt][r] *= corr;
        pw[(r + half * 8) * 32 + mla]      = (_Float16)p0;
        pw[(r + half * 8) * 32 + 16 + mla] = (_Float16)p1;
      }

      // P (C layout) -> A layout via per-wave LDS round-trip
      const _Float16* pa = pw + mla * 32;
      v16h pf = cat16(*(const v8h*)(pa + kb8), *(const v8h*)(pa + kb8 + 16));

      // O += P @ V  (8 N-tiles over HD)
#pragma unroll
      for (int nt = 0; nt < 8; ++nt) {
        const _Float16* pv = vlds + (nt * 16 + mla) * 32 + kb8;
        v16h bf = cat16(*(const v8h*)pv, *(const v8h*)(pv + 16));
        o[nt] = wmma_f16(pf, bf, o[nt]);
      }
    }
    __syncthreads();
  }

  // normalize + store f16 attention output as [B,S,H,HD]
#pragma unroll
  for (int r = 0; r < 8; ++r) {
    float inv = 1.0f / lI[r];
    int qrow = qw + r + half * 8;
#pragma unroll
    for (int nt = 0; nt < 8; ++nt) {
      int hd = nt * 16 + mla;
      oh[((size_t)b * SQ + qrow) * DQ + h * HDQ + hd] = (_Float16)(o[nt][r] * inv);
    }
  }
}

// ---------------------------------------------------------------------------
// Kernel 3: output projection attnout[8192,2048] @ wo^T -> fp32 d_out
// ---------------------------------------------------------------------------
__global__ __launch_bounds__(256) void proj_kernel(
    const _Float16* __restrict__ ah,    // [B*S, D]
    const _Float16* __restrict__ wh,    // [D, D] (N x K)
    float* __restrict__ out)
{
  __shared__ __align__(16) _Float16 As[128 * 32];
  __shared__ __align__(16) _Float16 Bs[128 * 32];
  const int t = threadIdx.x, lane = t & 31, w = t >> 5;
  const int wm = w >> 2, wn = w & 3;
  const int m0 = blockIdx.y * 128, n0 = blockIdx.x * 128;
  const int mla = lane & 15, half = lane >> 4;
  const int kb8 = half * 8, koff16 = half * 16;
  const int arow = t >> 1, acol = (t & 1) * 16;

  v8f acc[4][2] = {};
  for (int kt = 0; kt < DQ; kt += 32) {
    const _Float16* gp = ah + (size_t)(m0 + arow) * DQ + kt + acol;
    async_ld16(As + arow * 32 + acol,     gp);
    async_ld16(As + arow * 32 + acol + 8, gp + 8);
    const _Float16* gq = wh + (size_t)(n0 + arow) * DQ + kt + acol;
    async_ld16(Bs + arow * 32 + acol,     gq);
    async_ld16(Bs + arow * 32 + acol + 8, gq + 8);
    if (kt + 32 < DQ) {
      __builtin_prefetch(gp + 32, 0, 1);
      __builtin_prefetch(gq + 32, 0, 1);
    }
    wait_async0();
    __syncthreads();

    v16h af[4], bf[2];
#pragma unroll
    for (int i = 0; i < 4; ++i) {
      const _Float16* p = As + (wm * 64 + i * 16 + mla) * 32;
      af[i] = cat16(*(const v8h*)(p + kb8), *(const v8h*)(p + kb8 + 16));
    }
#pragma unroll
    for (int j = 0; j < 2; ++j) {
      const _Float16* p = Bs + (wn * 32 + j * 16 + mla) * 32 + kb8;
      bf[j] = cat16(*(const v8h*)p, *(const v8h*)(p + 16));
    }
#pragma unroll
    for (int i = 0; i < 4; ++i)
#pragma unroll
      for (int j = 0; j < 2; ++j)
        acc[i][j] = wmma_f16(af[i], bf[j], acc[i][j]);
    __syncthreads();
  }

#pragma unroll
  for (int i = 0; i < 4; ++i)
#pragma unroll
    for (int j = 0; j < 2; ++j)
#pragma unroll
      for (int r = 0; r < 8; ++r) {
        int mg = m0 + wm * 64 + i * 16 + r + half * 8;
        int ng = n0 + wn * 32 + j * 16 + mla;
        out[(size_t)mg * DQ + ng] = acc[i][j][r];
      }
}

// ---------------------------------------------------------------------------
extern "C" void kernel_launch(void* const* d_in, const int* in_sizes, int n_in,
                              void* d_out, int out_size, void* d_ws, size_t ws_size,
                              hipStream_t stream) {
  (void)in_sizes; (void)n_in; (void)out_size; (void)ws_size;
  const float* x    = (const float*)d_in[0];
  const float* wqkv = (const float*)d_in[1];
  const float* wo   = (const float*)d_in[2];
  float* out = (float*)d_out;

  char* ws = (char*)d_ws;
  size_t off = 0;
  auto alloc = [&](size_t bytes) -> char* {
    char* p = ws + off;
    off += (bytes + 255) & ~(size_t)255;
    return p;
  };
  const size_t ND = (size_t)BQ * SQ * DQ;      // 16,777,216
  _Float16* xh  = (_Float16*)alloc(ND * 2);
  _Float16* wqh = (_Float16*)alloc((size_t)3 * DQ * DQ * 2);
  _Float16* woh = (_Float16*)alloc((size_t)DQ * DQ * 2);
  _Float16* qhp = (_Float16*)alloc(ND * 2);
  _Float16* khp = (_Float16*)alloc(ND * 2);
  _Float16* vth = (_Float16*)alloc(ND * 2);
  _Float16* aoh = (_Float16*)alloc(ND * 2);

  cast_f32_to_f16<<<4096, 256, 0, stream>>>(x, xh, (int)(ND / 8));
  cast_f32_to_f16<<<4096, 256, 0, stream>>>(wqkv, wqh, (3 * DQ * DQ) / 8);
  cast_f32_to_f16<<<2048, 256, 0, stream>>>(wo, woh, (DQ * DQ) / 8);

  // QKV + RoPE: M tiles = 8192/128 = 64, N tiles = 6144/128 = 48
  qkv_rope_kernel<<<dim3(48, 64), 256, 0, stream>>>(xh, wqh, qhp, khp, vth);

  // attention: x = query tile (S/128 = 16), y = b*H (64)
  attn_kernel<<<dim3(16, 64), 256, 0, stream>>>(qhp, khp, vth, aoh);

  // projection: N tiles = 2048/128 = 16, M tiles = 64
  proj_kernel<<<dim3(16, 64), 256, 0, stream>>>(aoh, woh, out);
}
